// CPUBlock_2465311228263
// MI455X (gfx1250) — compile-verified, matrix operand order checked by reading, not run on hardware
//
#include <hip/hip_runtime.h>
#include <cstdint>
#include <cstddef>

#define DEV __device__ __forceinline__

typedef __attribute__((ext_vector_type(16))) __bf16 v16bf;
typedef __attribute__((ext_vector_type(8)))  float  v8f;

namespace cfg {
constexpr int BATCH = 4, S = 1024, D = 1024, H = 16, HD = 64, TILES = 8, DFF = 4096;
constexpr int NTOK = BATCH * S;        // 4096 tokens
constexpr int D3   = 3 * D;            // 3072
constexpr int MT   = 64;               // GEMM macro-tile M (expert pad granularity)
constexpr int MAXT = 72;               // max 64-row m-tiles after per-expert padding
constexpr int CAP  = MAXT * MT;        // 4608 padded token slots
}

// ---------- bf16 helpers (explicit RNE, no reliance on __bf16 arithmetic) ----------
DEV __bf16 f2bf(float f) {
  unsigned u = __builtin_bit_cast(unsigned, f);
  unsigned r = (u + 0x7fffu + ((u >> 16) & 1u)) >> 16;
  unsigned short s = (unsigned short)r;
  return __builtin_bit_cast(__bf16, s);
}
DEV float bf2f(__bf16 b) {
  unsigned short s = __builtin_bit_cast(unsigned short, b);
  unsigned u = (unsigned)s << 16;
  return __builtin_bit_cast(float, u);
}

// ---------- WMMA fragment load from row-major LDS tile ----------
// ISA 7.12.2 16-bit operand layout: lanes 0-15 hold K {kb..kb+7, kb+16..kb+23} with
// kb=0; lanes 16-31 with kb=8. Row (A) / column (B) index = lane & 15.
DEV v16bf load_frag(const __bf16* lds, int row_base, int ld, int lane) {
  int r  = row_base + (lane & 15);
  int kb = (lane >> 4) * 8;
  const __bf16* p = lds + r * ld + kb;
  v16bf f;
#pragma unroll
  for (int j = 0; j < 8; ++j) { f[j] = p[j]; f[j + 8] = p[j + 16]; }
  return f;
}

DEV v8f wmma_bf16(v16bf a, v16bf b, v8f c) {
  return __builtin_amdgcn_wmma_f32_16x16x32_bf16(false, a, false, b, (short)0, c,
                                                 false, false);
}

// ---------- weight conversion kernels ----------
__global__ __launch_bounds__(256) void cvt_kernel(const float* __restrict__ src,
                                                  __bf16* __restrict__ dst, int n) {
  int i = blockIdx.x * 256 + threadIdx.x;
  if (i < n) dst[i] = f2bf(src[i]);
}

// src [E][K][Nn] f32 -> dst [E][Nn][K] bf16 (per-expert transpose)
__global__ __launch_bounds__(256) void cvtT_kernel(const float* __restrict__ src,
                                                   __bf16* __restrict__ dst,
                                                   int K, int Nn) {
  size_t i   = (size_t)blockIdx.x * 256 + threadIdx.x;
  size_t per = (size_t)K * Nn;
  size_t tot = per * cfg::TILES;
  if (i >= tot) return;
  size_t e = i / per, rem = i % per;
  size_t k = rem / Nn, n = rem % Nn;
  dst[e * per + n * K + k] = f2bf(src[i]);
}

// ---------- LayerNorm (f32 in -> bf16 out), one block per token ----------
__global__ __launch_bounds__(256) void ln_kernel(const float* __restrict__ x,
                                                 const float* __restrict__ g,
                                                 const float* __restrict__ b,
                                                 __bf16* __restrict__ out) {
  const int row = blockIdx.x;
  const float* xr = x + (size_t)row * cfg::D;
  float s = 0.f, s2 = 0.f;
  for (int i = threadIdx.x; i < cfg::D; i += 256) { float v = xr[i]; s += v; s2 += v * v; }
  __shared__ float r1[256], r2[256];
  r1[threadIdx.x] = s; r2[threadIdx.x] = s2;
  __syncthreads();
  for (int k = 128; k > 0; k >>= 1) {
    if (threadIdx.x < k) { r1[threadIdx.x] += r1[threadIdx.x + k];
                           r2[threadIdx.x] += r2[threadIdx.x + k]; }
    __syncthreads();
  }
  float mean = r1[0] * (1.0f / cfg::D);
  float var  = r2[0] * (1.0f / cfg::D) - mean * mean;
  float rs   = rsqrtf(var + 1e-5f);
  for (int i = threadIdx.x; i < cfg::D; i += 256)
    out[(size_t)row * cfg::D + i] = f2bf((xr[i] - mean) * rs * g[i] + b[i]);
}

// ---------- generic bf16 WMMA GEMM: 64 x 256 macro-tile per block ----------
// 8 waves = 4 M-subtiles x 2 N-halves; 8 wmma per wave per K-step.
// W slab (256x32 bf16) is streamed global->LDS with GLOBAL_LOAD_ASYNC_TO_LDS_B128
// (ASYNCcnt), A tile uses the regular path (needs gather + zero-fill).
enum { GF_RELU = 1, GF_RES = 2, GF_OUTBF = 4, GF_GATHER = 8, GF_SCATTER = 16, GF_EXPERT = 32 };

template <int FLAGS>
__global__ __launch_bounds__(256) void gemm64_kernel(
    const __bf16* __restrict__ A, int lda,
    const __bf16* __restrict__ W, size_t w_estride,   // W stored [Ntot][K] bf16
    const float* __restrict__ bias, int b_estride,
    float* __restrict__ outF, __bf16* __restrict__ outB, int ldo,
    const float* __restrict__ res,
    const int* __restrict__ perm,
    const int* __restrict__ tile_expert,
    const int* __restrict__ total_tiles,
    int K) {
  if (total_tiles && (int)blockIdx.x >= *total_tiles) return;
  const int tid = threadIdx.x, lane = tid & 31, wave = tid >> 5;
  const int msub = wave >> 1;                 // 0..3 -> rows [msub*16, +16)
  const int nh   = wave & 1;                  // 0..1 -> cols [nh*128, +128)
  const int mbase = blockIdx.x * cfg::MT;
  const int nb    = blockIdx.y * 256;
  int e = 0;
  if (FLAGS & GF_EXPERT) e = tile_expert[blockIdx.x];
  const __bf16* Wb    = W + (size_t)e * w_estride;
  const float*  biasE = bias + (size_t)e * b_estride;

  __shared__ __bf16 sA[64 * 32];
  __shared__ __bf16 sW[256 * 32];

  v8f acc[8] = {};

  // generic->LDS: low 32 bits of a flat pointer into LDS is the DS byte address
  const uint32_t ldsW = (uint32_t)(uintptr_t)(&sW[tid * 32]);

  int arow = mbase + (tid >> 2);
  if (FLAGS & GF_GATHER) arow = perm[arow];
  const int acol = (tid & 3) * 8;

  for (int k0 = 0; k0 < K; k0 += 32) {
    // async DMA of one 256x32 bf16 weight slab straight into LDS (64B per thread)
    {
      uint64_t gaddr = (uint64_t)(uintptr_t)(Wb + (size_t)(nb + tid) * K + k0);
      asm volatile(
          "global_load_async_to_lds_b128 %0, %1, off\n\t"
          "global_load_async_to_lds_b128 %0, %1, off offset:16\n\t"
          "global_load_async_to_lds_b128 %0, %1, off offset:32\n\t"
          "global_load_async_to_lds_b128 %0, %1, off offset:48"
          :: "v"(ldsW), "v"(gaddr) : "memory");
    }
    // A tile 64x32 (optionally gathered through perm; -1 rows are zero)
    {
      const __bf16 z = f2bf(0.f);
      if (arow >= 0 && k0 + 64 <= K)
        __builtin_prefetch(&A[(size_t)arow * lda + k0 + 32 + acol], 0, 1);
#pragma unroll
      for (int j = 0; j < 8; ++j)
        sA[(tid >> 2) * 32 + acol + j] =
            (arow < 0) ? z : A[(size_t)arow * lda + k0 + acol + j];
    }
    asm volatile("s_wait_asynccnt 0" ::: "memory");
    __syncthreads();
    v16bf a = load_frag(sA, msub * 16, 32, lane);
#pragma unroll
    for (int t = 0; t < 8; ++t) {
      v16bf b = load_frag(sW, nh * 128 + t * 16, 32, lane);
      acc[t] = wmma_bf16(a, b, acc[t]);
    }
    __syncthreads();
  }

  const int g = lane >> 4, lnn = lane & 15;
#pragma unroll
  for (int t = 0; t < 8; ++t)
#pragma unroll
    for (int i = 0; i < 8; ++i) {
      int m    = msub * 16 + i + 8 * g;
      int colN = nb + nh * 128 + t * 16 + lnn;
      float v  = acc[t][i] + biasE[colN];
      if (FLAGS & GF_RELU) v = v > 0.f ? v : 0.f;
      int orow = mbase + m;
      if (FLAGS & GF_SCATTER) { orow = perm[orow]; if (orow < 0) continue; }
      size_t oidx = (size_t)orow * ldo + colN;
      if (FLAGS & GF_RES) v += res[oidx];
      if (FLAGS & GF_OUTBF) outB[oidx] = f2bf(v);
      else                  outF[oidx] = v;
    }
}

// ---------- flash attention: one wave per (b, h, 16 query rows) ----------
__global__ __launch_bounds__(32) void attn_kernel(const __bf16* __restrict__ qkv,
                                                  __bf16* __restrict__ attn_out) {
  const int lane = threadIdx.x & 31;
  const int qt = blockIdx.x & 63;            // S/16 = 64
  const int h  = (blockIdx.x >> 6) & 15;
  const int b  = blockIdx.x >> 10;
  const int g = lane >> 4, lnn = lane & 15;
  const size_t rs = cfg::D3;
  const size_t base = (size_t)b * cfg::S * rs;
  const int kb = g * 8;

  // Q A-fragments over hd=64 -> two K=32 halves
  const __bf16* qp = qkv + base + (size_t)(qt * 16 + lnn) * rs + h * cfg::HD;
  v16bf qf[2];
#pragma unroll
  for (int hh = 0; hh < 2; ++hh)
#pragma unroll
    for (int j = 0; j < 8; ++j) {
      qf[hh][j]     = qp[hh * 32 + kb + j];
      qf[hh][j + 8] = qp[hh * 32 + kb + 16 + j];
    }

  v8f o[4] = {};
  float mrun[8], lrun[8];
#pragma unroll
  for (int i = 0; i < 8; ++i) { mrun[i] = -3.0e38f; lrun[i] = 0.f; }

  __shared__ __bf16 sP[16 * 32];

  for (int c = 0; c < cfg::S / 32; ++c) {
    // scores: Q[16x64] x K^T[64x32] -> two 16x16 tiles
    v8f sc[2] = {};
#pragma unroll
    for (int nt = 0; nt < 2; ++nt) {
      const __bf16* kp =
          qkv + base + (size_t)(c * 32 + nt * 16 + lnn) * rs + cfg::D + h * cfg::HD;
#pragma unroll
      for (int hh = 0; hh < 2; ++hh) {
        v16bf kf;
#pragma unroll
        for (int j = 0; j < 8; ++j) {
          kf[j]     = kp[hh * 32 + kb + j];
          kf[j + 8] = kp[hh * 32 + kb + 16 + j];
        }
        sc[nt] = wmma_bf16(qf[hh], kf, sc[nt]);
      }
    }
#pragma unroll
    for (int nt = 0; nt < 2; ++nt)
#pragma unroll
      for (int i = 0; i < 8; ++i) sc[nt][i] *= 0.125f;   // 1/sqrt(64)

    // online softmax (row reductions across 16-lane groups)
    float nm[8], alpha[8], ls[8];
#pragma unroll
    for (int i = 0; i < 8; ++i) {
      float mx = fmaxf(sc[0][i], sc[1][i]);
#pragma unroll
      for (int m = 1; m < 16; m <<= 1) mx = fmaxf(mx, __shfl_xor(mx, m, 32));
      nm[i] = fmaxf(mrun[i], mx);
      alpha[i] = __expf(mrun[i] - nm[i]);
      mrun[i] = nm[i];
      ls[i] = 0.f;
    }
#pragma unroll
    for (int nt = 0; nt < 2; ++nt)
#pragma unroll
      for (int i = 0; i < 8; ++i) {
        float p = __expf(sc[nt][i] - nm[i]);
        ls[i] += p;
        sP[(i + 8 * g) * 32 + nt * 16 + lnn] = f2bf(p);
      }
#pragma unroll
    for (int i = 0; i < 8; ++i) {
#pragma unroll
      for (int m = 1; m < 16; m <<= 1) ls[i] += __shfl_xor(ls[i], m, 32);
      lrun[i] = lrun[i] * alpha[i] + ls[i];
    }
#pragma unroll
    for (int t = 0; t < 4; ++t)
#pragma unroll
      for (int i = 0; i < 8; ++i) o[t][i] *= alpha[i];

    __syncthreads();
    v16bf pa = load_frag(sP, 0, 32, lane);   // re-swizzle P into A layout
#pragma unroll
    for (int t = 0; t < 4; ++t) {
      v16bf vf;
      int dcol = t * 16 + lnn;
#pragma unroll
      for (int j = 0; j < 8; ++j) {
        size_t r0 = base + (size_t)(c * 32 + kb + j) * rs      + 2 * cfg::D + h * cfg::HD;
        size_t r1 = base + (size_t)(c * 32 + kb + 16 + j) * rs + 2 * cfg::D + h * cfg::HD;
        vf[j]     = qkv[r0 + dcol];
        vf[j + 8] = qkv[r1 + dcol];
      }
      o[t] = wmma_bf16(pa, vf, o[t]);
    }
    __syncthreads();
  }

#pragma unroll
  for (int t = 0; t < 4; ++t)
#pragma unroll
    for (int i = 0; i < 8; ++i) {
      int row = qt * 16 + i + 8 * g;
      int col = h * cfg::HD + t * 16 + lnn;
      attn_out[((size_t)b * cfg::S + row) * cfg::D + col] = f2bf(o[t][i] / lrun[i]);
    }
}

// ---------- MoE gate / bucketing ----------
__global__ __launch_bounds__(32) void zero8_kernel(int* counts) {
  if (threadIdx.x < cfg::TILES) counts[threadIdx.x] = 0;
}

__global__ __launch_bounds__(256) void gate_kernel(const __bf16* __restrict__ n2,
                                                   const float* __restrict__ gw,
                                                   const float* __restrict__ gb,
                                                   float* __restrict__ gate_out,
                                                   int* __restrict__ expert_id,
                                                   int* __restrict__ counts) {
  int wave = threadIdx.x >> 5, lane = threadIdx.x & 31;
  int tok = blockIdx.x * 8 + wave;
  float acc[cfg::TILES];
#pragma unroll
  for (int e = 0; e < cfg::TILES; ++e) acc[e] = 0.f;
  for (int k = lane; k < cfg::D; k += 32) {
    float tv = bf2f(n2[(size_t)tok * cfg::D + k]);
#pragma unroll
    for (int e = 0; e < cfg::TILES; ++e) acc[e] += tv * gw[e * cfg::D + k];
  }
#pragma unroll
  for (int e = 0; e < cfg::TILES; ++e)
#pragma unroll
    for (int m = 16; m >= 1; m >>= 1) acc[e] += __shfl_xor(acc[e], m, 32);
  int best = 0; float bv = acc[0] + gb[0];
#pragma unroll
  for (int e = 1; e < cfg::TILES; ++e) {
    float v = acc[e] + gb[e];
    if (v > bv) { bv = v; best = e; }
  }
  if (lane < cfg::TILES) gate_out[(size_t)tok * cfg::TILES + lane] = (lane == best) ? 1.f : 0.f;
  if (lane == 0) { expert_id[tok] = best; atomicAdd(&counts[best], 1); }
}

__global__ __launch_bounds__(256) void plan_kernel(const int* __restrict__ counts,
                                                   int* __restrict__ cursors,
                                                   int* __restrict__ tile_expert,
                                                   int* __restrict__ total_tiles,
                                                   int* __restrict__ perm) {
  for (int i = threadIdx.x; i < cfg::CAP; i += 256) perm[i] = -1;
  if (threadIdx.x == 0) {
    int off = 0;
    for (int e = 0; e < cfg::TILES; ++e) {
      cursors[e] = off;
      int tiles = (counts[e] + cfg::MT - 1) >> 6;   // pad segments to 64 rows
      for (int t = 0; t < tiles; ++t) tile_expert[(off >> 6) + t] = e;
      off += tiles << 6;
    }
    *total_tiles = off >> 6;
  }
}

__global__ __launch_bounds__(256) void scatter_kernel(const int* __restrict__ expert_id,
                                                      int* __restrict__ cursors,
                                                      int* __restrict__ perm) {
  int n = blockIdx.x * 256 + threadIdx.x;
  int e = expert_id[n];
  int pos = atomicAdd(&cursors[e], 1);
  perm[pos] = n;
}

// ---------- host launcher ----------
extern "C" void kernel_launch(void* const* d_in, const int* in_sizes, int n_in,
                              void* d_out, int out_size, void* d_ws, size_t ws_size,
                              hipStream_t stream) {
  using namespace cfg;
  const float* x      = (const float*)d_in[0];
  const float* ln1_g  = (const float*)d_in[1];
  const float* ln1_b  = (const float*)d_in[2];
  const float* qkv_w  = (const float*)d_in[3];
  const float* qkv_b  = (const float*)d_in[4];
  const float* out_w  = (const float*)d_in[5];
  const float* out_b  = (const float*)d_in[6];
  const float* ln2_g  = (const float*)d_in[7];
  const float* ln2_b  = (const float*)d_in[8];
  const float* gate_w = (const float*)d_in[9];
  const float* gate_b = (const float*)d_in[10];
  const float* up_W   = (const float*)d_in[11];
  const float* up_b   = (const float*)d_in[12];
  const float* down_W = (const float*)d_in[13];
  const float* down_b = (const float*)d_in[14];

  float* out_main = (float*)d_out;                  // [NTOK, D]
  float* out_gate = out_main + (size_t)NTOK * D;    // [NTOK, TILES]

  char* ws = (char*)d_ws;
  size_t off = 0;
  auto take = [&](size_t bytes) -> char* {
    char* p = ws + off;
    off += (bytes + 255) & ~(size_t)255;
    return p;
  };
  __bf16* qkvw_bf = (__bf16*)take((size_t)D3 * D * 2);
  __bf16* outw_bf = (__bf16*)take((size_t)D * D * 2);
  __bf16* upT_bf  = (__bf16*)take((size_t)TILES * DFF * D * 2);
  __bf16* dnT_bf  = (__bf16*)take((size_t)TILES * D * DFF * 2);
  __bf16* n1_bf   = (__bf16*)take((size_t)NTOK * D * 2);
  __bf16* qkv_bf  = (__bf16*)take((size_t)NTOK * D3 * 2);
  __bf16* attn_bf = (__bf16*)take((size_t)NTOK * D * 2);
  float*  x1_f    = (float*)take((size_t)NTOK * D * 4);
  __bf16* n2_bf   = (__bf16*)take((size_t)NTOK * D * 2);
  __bf16* h_bf    = (__bf16*)take((size_t)CAP * DFF * 2);
  int* expert_id  = (int*)take((size_t)NTOK * 4);
  int* perm       = (int*)take((size_t)CAP * 4);
  int* counts     = (int*)take(8 * 4);
  int* cursors    = (int*)take(8 * 4);
  int* tile_exp   = (int*)take((size_t)MAXT * 4);
  int* total      = (int*)take(4);
  (void)off; (void)ws_size; (void)in_sizes; (void)n_in; (void)out_size;

  // 1) weight precision / layout conversion
  cvt_kernel<<<(D3 * D + 255) / 256, 256, 0, stream>>>(qkv_w, qkvw_bf, D3 * D);
  cvt_kernel<<<(D * D + 255) / 256, 256, 0, stream>>>(out_w, outw_bf, D * D);
  cvtT_kernel<<<(int)(((size_t)TILES * D * DFF + 255) / 256), 256, 0, stream>>>(
      up_W, upT_bf, D, DFF);          // [8][1024][4096] -> [8][4096][1024]
  cvtT_kernel<<<(int)(((size_t)TILES * DFF * D + 255) / 256), 256, 0, stream>>>(
      down_W, dnT_bf, DFF, D);        // [8][4096][1024] -> [8][1024][4096]

  // 2) LN1 -> QKV -> attention -> out-proj (+residual x)
  ln_kernel<<<NTOK, 256, 0, stream>>>(x, ln1_g, ln1_b, n1_bf);
  gemm64_kernel<GF_OUTBF><<<dim3(NTOK / MT, D3 / 256), 256, 0, stream>>>(
      n1_bf, D, qkvw_bf, 0, qkv_b, 0, nullptr, qkv_bf, D3,
      nullptr, nullptr, nullptr, nullptr, D);
  attn_kernel<<<BATCH * H * (S / 16), 32, 0, stream>>>(qkv_bf, attn_bf);
  gemm64_kernel<GF_RES><<<dim3(NTOK / MT, D / 256), 256, 0, stream>>>(
      attn_bf, D, outw_bf, 0, out_b, 0, x1_f, nullptr, D,
      x, nullptr, nullptr, nullptr, D);

  // 3) LN2 -> gate -> expert bucketing
  ln_kernel<<<NTOK, 256, 0, stream>>>(x1_f, ln2_g, ln2_b, n2_bf);
  zero8_kernel<<<1, 32, 0, stream>>>(counts);
  gate_kernel<<<NTOK / 8, 256, 0, stream>>>(n2_bf, gate_w, gate_b, out_gate,
                                            expert_id, counts);
  plan_kernel<<<1, 256, 0, stream>>>(counts, cursors, tile_exp, total, perm);
  scatter_kernel<<<NTOK / 256, 256, 0, stream>>>(expert_id, cursors, perm);

  // 4) grouped expert GEMMs: up (gather + ReLU, bf16 h) then down (scatter + residual)
  gemm64_kernel<GF_RELU | GF_OUTBF | GF_GATHER | GF_EXPERT>
      <<<dim3(MAXT, DFF / 256), 256, 0, stream>>>(
          n2_bf, D, upT_bf, (size_t)DFF * D, up_b, DFF, nullptr, h_bf, DFF,
          nullptr, perm, tile_exp, total, D);
  gemm64_kernel<GF_RES | GF_SCATTER | GF_EXPERT>
      <<<dim3(MAXT, D / 256), 256, 0, stream>>>(
          h_bf, DFF, dnT_bf, (size_t)D * DFF, down_b, D, out_main, nullptr, D,
          x1_f, perm, tile_exp, total, DFF);
}
